// MultiHeadedAttention_65326452572415
// MI455X (gfx1250) — matrix operand order repaired and checked by reading, not compiled, not run on hardware
//
#include <hip/hip_runtime.h>
#include <hip/hip_bf16.h>

// MHA: B=4, S=2048, N_UNITS=1024, H=16, Dk=64. All matmuls on v_wmma_f32_16x16x32_bf16.
// Pipeline: [f32->bf16 cvt + mask bitpack] -> 3x proj GEMM (Q pre-scaled by 0.125*log2e,
// V stored transposed) -> flash attn (base-2 softmax, row-sums via ones-WMMA) -> out GEMM.

typedef __bf16 v16bf __attribute__((ext_vector_type(16)));
typedef __bf16 v8bf  __attribute__((ext_vector_type(8)));
typedef float  v8f   __attribute__((ext_vector_type(8)));

#define S_LEN 2048
#define NU    1024
#define NHEAD 16
#define DK    64
#define MROWS 8192   // B*S
#define SW    64     // S/32 mask words per row
#define LOG2E 1.44269504088896340736f

// ---------------------------------------------------------------------------
// One-shot f32 -> bf16 conversion (8 elements/thread).
// ---------------------------------------------------------------------------
__global__ __launch_bounds__(256) void cvt_f32_to_bf16(
    const float* __restrict__ src, __bf16* __restrict__ dst, int n8)
{
    const int i = blockIdx.x * blockDim.x + threadIdx.x;
    if (i >= n8) return;
    const float4* s = (const float4*)src;
    const float4 a = s[2 * i], b = s[2 * i + 1];
    v8bf o;
    o[0] = (__bf16)a.x; o[1] = (__bf16)a.y; o[2] = (__bf16)a.z; o[3] = (__bf16)a.w;
    o[4] = (__bf16)b.x; o[5] = (__bf16)b.y; o[6] = (__bf16)b.z; o[7] = (__bf16)b.w;
    *(v8bf*)(dst + 8 * (size_t)i) = o;
}

// ---------------------------------------------------------------------------
// Pack int32 mask -> 1 bit per element (64 MB -> 2 MB; 32x less attn traffic).
// ---------------------------------------------------------------------------
__global__ __launch_bounds__(256) void pack_mask_bits(
    const int* __restrict__ m, unsigned* __restrict__ out, int nwords)
{
    const int w = blockIdx.x * blockDim.x + threadIdx.x;
    if (w >= nwords) return;
    const int4* p = (const int4*)(m + (size_t)w * 32);
    unsigned acc = 0;
    #pragma unroll
    for (int j = 0; j < 8; ++j) {
        const int4 v = p[j];
        acc |= (unsigned)(v.x != 0) << (4 * j);
        acc |= (unsigned)(v.y != 0) << (4 * j + 1);
        acc |= (unsigned)(v.z != 0) << (4 * j + 2);
        acc |= (unsigned)(v.w != 0) << (4 * j + 3);
    }
    out[w] = acc;
}

__device__ __forceinline__ v16bf cat8(v8bf lo, v8bf hi) {
    return __builtin_shufflevector(lo, hi, 0, 1, 2, 3, 4, 5, 6, 7,
                                           8, 9, 10, 11, 12, 13, 14, 15);
}

// ---------------------------------------------------------------------------
// Projection GEMM (all-bf16): P = ((X @ W^T) + bias) * oscale.
//   VT=false: dst[b,h,s,d]   (Q with oscale=0.125*log2e, K with oscale=1)
//   VT=true : dst[b,h,d,s]   (V transposed, oscale=1)
// One wave -> 16x64 tile, K-loop step 32.
// ---------------------------------------------------------------------------
template <bool VT>
__global__ __launch_bounds__(256) void mha_proj(
    const __bf16* __restrict__ Xb, const __bf16* __restrict__ Wb,
    const float* __restrict__ bias, __bf16* __restrict__ dst, float oscale)
{
    const int lane = threadIdx.x & 31, wvid = threadIdx.x >> 5;
    const int hf = lane >> 4, r16 = lane & 15;
    const int tile = blockIdx.x * 8 + wvid;
    const int m0 = (tile >> 4) * 16;
    const int n0 = (tile & 15) * 64;

    v8f acc[4] = {};
    const __bf16* arow = Xb + (size_t)(m0 + r16) * NU;

    for (int kb = 0; kb < NU; kb += 32) {
        const v16bf a = *(const v16bf*)(arow + kb + hf * 16);
        #pragma unroll
        for (int t = 0; t < 4; ++t) {
            const v16bf bb = *(const v16bf*)(Wb + (size_t)(n0 + t * 16 + r16) * NU + kb + hf * 16);
            acc[t] = __builtin_amdgcn_wmma_f32_16x16x32_bf16(
                false, a, false, bb, (short)0, acc[t], false, false);
        }
    }
    #pragma unroll
    for (int t = 0; t < 4; ++t) {
        const int n = n0 + t * 16 + r16;
        const int h = n >> 6, d = n & 63;
        const float bn = bias[n];
        #pragma unroll
        for (int r = 0; r < 8; ++r) {
            const int m = m0 + r + 8 * hf;
            const int b = m >> 11, s = m & (S_LEN - 1);
            const size_t bh = (size_t)b * NHEAD + h;
            const size_t idx = VT ? (bh * DK + d) * S_LEN + s
                                  : (bh * S_LEN + s) * DK + d;
            dst[idx] = (__bf16)((acc[t][r] + bn) * oscale);
        }
    }
}

// ---------------------------------------------------------------------------
// Flash attention: one wave per (b,h, 32-query tile) = two 16-row q-groups.
// Q pre-scaled by 0.125*log2e -> scores come out of WMMA ready for exp2.
// Row sums computed by an extra WMMA against an all-ones B fragment (same
// C-layout as Hacc -> rescales identically); only row-max uses shuffles.
// ---------------------------------------------------------------------------
__global__ __launch_bounds__(128) void mha_flash_attn(
    const __bf16* __restrict__ Qw, const __bf16* __restrict__ Kw,
    const __bf16* __restrict__ Vt, const unsigned* __restrict__ Mp,
    __bf16* __restrict__ Hw)
{
    __shared__ __attribute__((aligned(16))) __bf16 smem[4 * 2 * 16 * 40]; // stride 40: conflict-free
    const int wvid = threadIdx.x >> 5, lane = threadIdx.x & 31;
    const int hf = lane >> 4, r16 = lane & 15;
    const int gw = blockIdx.x * 4 + wvid;        // 0..4095
    const int qt = gw & 63;                      // S/32 query tiles
    const int bh = gw >> 6;                      // b*H + h
    const int b = bh >> 4, h = bh & 15;
    const int qbase = qt * 32;

    const __bf16* Qb = Qw + (size_t)bh * S_LEN * DK;
    const __bf16* Kb = Kw + (size_t)bh * S_LEN * DK;
    const __bf16* Vb = Vt + (size_t)bh * DK * S_LEN;        // [d][s]
    const unsigned* mrow = Mp + (size_t)b * S_LEN * SW;     // bit per key

    v16bf aQ[2][2];
    #pragma unroll
    for (int g = 0; g < 2; ++g) {
        const __bf16* qrow = Qb + (size_t)(qbase + g * 16 + r16) * DK;
        aQ[g][0] = *(const v16bf*)(qrow + hf * 16);
        aQ[g][1] = *(const v16bf*)(qrow + 32 + hf * 16);
    }

    v16bf bOnes;
    #pragma unroll
    for (int j = 0; j < 16; ++j) bOnes[j] = (__bf16)1.0f;

    v8f Hacc[2][4] = {};
    v8f Lacc[2] = {};                            // row sums via ones-WMMA
    float m_run[2][8];
    #pragma unroll
    for (int g = 0; g < 2; ++g)
        #pragma unroll
        for (int r = 0; r < 8; ++r) m_run[g][r] = -1e30f;

    for (int kb = 0; kb < S_LEN; kb += 32) {
        // ---- shared K / V fragments for this 32-key block ----
        v16bf bK[2][2];
        #pragma unroll
        for (int kt = 0; kt < 2; ++kt) {
            const __bf16* krow = Kb + (size_t)(kb + kt * 16 + r16) * DK;
            bK[kt][0] = *(const v16bf*)(krow + hf * 16);
            bK[kt][1] = *(const v16bf*)(krow + 32 + hf * 16);
        }
        v16bf bV[4];
        #pragma unroll
        for (int t = 0; t < 4; ++t)
            bV[t] = *(const v16bf*)(Vb + (size_t)(t * 16 + r16) * S_LEN + kb + hf * 16);

        // ---- mask words: one u32 covers the 32-key block for one q-row ----
        unsigned mw[2][8];
        #pragma unroll
        for (int g = 0; g < 2; ++g)
            #pragma unroll
            for (int r = 0; r < 8; ++r)
                mw[g][r] = mrow[(size_t)(qbase + g * 16 + r + 8 * hf) * SW + (kb >> 5)];

        #pragma unroll
        for (int g = 0; g < 2; ++g) {
            // ---- scores (already scaled by 0.125*log2e via Q) ----
            v8f sc[2];
            #pragma unroll
            for (int kt = 0; kt < 2; ++kt) {
                v8f z = {};
                z = __builtin_amdgcn_wmma_f32_16x16x32_bf16(false, aQ[g][0], false, bK[kt][0], (short)0, z, false, false);
                sc[kt] = __builtin_amdgcn_wmma_f32_16x16x32_bf16(false, aQ[g][1], false, bK[kt][1], (short)0, z, false, false);
            }
            // ---- mask bits ----
            #pragma unroll
            for (int kt = 0; kt < 2; ++kt)
                #pragma unroll
                for (int r = 0; r < 8; ++r)
                    if (((mw[g][r] >> (kt * 16 + r16)) & 1u) == 0u) sc[kt][r] = -1e9f;

            // ---- online softmax (base 2): only row-max needs shuffles ----
            float scl[8];
            #pragma unroll
            for (int r = 0; r < 8; ++r) {
                float rmax = fmaxf(sc[0][r], sc[1][r]);
                #pragma unroll
                for (int off = 1; off < 16; off <<= 1)
                    rmax = fmaxf(rmax, __shfl_xor(rmax, off, 32));
                const float mnew = fmaxf(m_run[g][r], rmax);
                scl[r] = exp2f(m_run[g][r] - mnew);
                sc[0][r] = exp2f(sc[0][r] - mnew);
                sc[1][r] = exp2f(sc[1][r] - mnew);
                m_run[g][r] = mnew;
            }
            #pragma unroll
            for (int t = 0; t < 4; ++t)
                #pragma unroll
                for (int r = 0; r < 8; ++r) Hacc[g][t][r] *= scl[r];
            #pragma unroll
            for (int r = 0; r < 8; ++r) Lacc[g][r] *= scl[r];

            // ---- probs: C-layout -> LDS -> A-fragment layout ----
            __bf16* wl = smem + wvid * 1280 + g * 640;
            #pragma unroll
            for (int kt = 0; kt < 2; ++kt)
                #pragma unroll
                for (int r = 0; r < 8; ++r)
                    wl[(r + 8 * hf) * 40 + kt * 16 + r16] = (__bf16)sc[kt][r];

            const v16bf aP = cat8(*(const v8bf*)(wl + r16 * 40 + hf * 16),
                                  *(const v8bf*)(wl + r16 * 40 + hf * 16 + 8));

            // ---- row sums: L += P x ones (same C-layout as Hacc) ----
            Lacc[g] = __builtin_amdgcn_wmma_f32_16x16x32_bf16(
                false, aP, false, bOnes, (short)0, Lacc[g], false, false);

            // ---- AV: shared bV fragments ----
            #pragma unroll
            for (int t = 0; t < 4; ++t)
                Hacc[g][t] = __builtin_amdgcn_wmma_f32_16x16x32_bf16(
                    false, aP, false, bV[t], (short)0, Hacc[g][t], false, false);
        }
    }

    // ---- epilogue: H[b,s,h*64+d] bf16, normalize by row sums ----
    #pragma unroll
    for (int g = 0; g < 2; ++g)
        #pragma unroll
        for (int r = 0; r < 8; ++r) {
            const int q = qbase + g * 16 + r + 8 * hf;
            const float inv = 1.0f / Lacc[g][r];
            #pragma unroll
            for (int t = 0; t < 4; ++t)
                Hw[((size_t)b * S_LEN + q) * NU + h * DK + t * 16 + r16] =
                    (__bf16)(Hacc[g][t][r] * inv);
        }
}

// ---------------------------------------------------------------------------
// Output projection: out f32 [MROWS,NU] = Hbf16 @ Wo^T + bo
// ---------------------------------------------------------------------------
__global__ __launch_bounds__(256) void mha_out_proj(
    const __bf16* __restrict__ Hb, const __bf16* __restrict__ Wb,
    const float* __restrict__ bias, float* __restrict__ out)
{
    const int lane = threadIdx.x & 31, wvid = threadIdx.x >> 5;
    const int hf = lane >> 4, r16 = lane & 15;
    const int tile = blockIdx.x * 8 + wvid;
    const int m0 = (tile >> 4) * 16;
    const int n0 = (tile & 15) * 64;

    v8f acc[4] = {};
    const __bf16* arow = Hb + (size_t)(m0 + r16) * NU;

    for (int kb = 0; kb < NU; kb += 32) {
        const v16bf a = *(const v16bf*)(arow + kb + hf * 16);
        #pragma unroll
        for (int t = 0; t < 4; ++t) {
            const v16bf bb = *(const v16bf*)(Wb + (size_t)(n0 + t * 16 + r16) * NU + kb + hf * 16);
            acc[t] = __builtin_amdgcn_wmma_f32_16x16x32_bf16(
                false, a, false, bb, (short)0, acc[t], false, false);
        }
    }
    #pragma unroll
    for (int t = 0; t < 4; ++t) {
        const int n = n0 + t * 16 + r16;
        const float bn = bias[n];
        #pragma unroll
        for (int r = 0; r < 8; ++r) {
            const int m = m0 + r + 8 * hf;
            out[(size_t)m * NU + n] = acc[t][r] + bn;
        }
    }
}

extern "C" void kernel_launch(void* const* d_in, const int* in_sizes, int n_in,
                              void* d_out, int out_size, void* d_ws, size_t ws_size,
                              hipStream_t stream) {
    (void)in_sizes; (void)n_in; (void)out_size; (void)ws_size;
    const float* query = (const float*)d_in[0];
    const float* key   = (const float*)d_in[1];
    const float* value = (const float*)d_in[2];
    const int*   mask  = (const int*)d_in[3];
    const float* Wq = (const float*)d_in[4];  const float* bq = (const float*)d_in[5];
    const float* Wk = (const float*)d_in[6];  const float* bk = (const float*)d_in[7];
    const float* Wv = (const float*)d_in[8];  const float* bv = (const float*)d_in[9];
    const float* Wo = (const float*)d_in[10]; const float* bo = (const float*)d_in[11];

    // Workspace: 6 activation segs + 4 weight segs (bf16) + 2MB packed mask = ~106 MiB.
    __bf16* wsb = (__bf16*)d_ws;
    const size_t seg = (size_t)MROWS * NU;       // 8,388,608
    const size_t wseg = (size_t)NU * NU;         // 1,048,576
    __bf16* Xqb = wsb;                            // later reused as H
    __bf16* Xkb = wsb + seg;
    __bf16* Xvb = wsb + 2 * seg;
    __bf16* Qb  = wsb + 3 * seg;
    __bf16* Kb  = wsb + 4 * seg;
    __bf16* Vt  = wsb + 5 * seg;
    __bf16* Wqb = wsb + 6 * seg;
    __bf16* Wkb = Wqb + wseg;
    __bf16* Wvb = Wkb + wseg;
    __bf16* Wob = Wvb + wseg;
    unsigned* Mp = (unsigned*)(Wob + wseg);       // 524,288 words (2 MiB)
    __bf16* Hb  = Xqb;                            // query copy dead after proj Q

    // 1) one-shot conversions + mask bit-pack (bandwidth pass)
    const int x8 = (int)(seg / 8), w8 = (int)(wseg / 8);
    cvt_f32_to_bf16<<<x8 / 256, 256, 0, stream>>>(query, Xqb, x8);
    cvt_f32_to_bf16<<<x8 / 256, 256, 0, stream>>>(key,   Xkb, x8);
    cvt_f32_to_bf16<<<x8 / 256, 256, 0, stream>>>(value, Xvb, x8);
    cvt_f32_to_bf16<<<w8 / 256, 256, 0, stream>>>(Wq, Wqb, w8);
    cvt_f32_to_bf16<<<w8 / 256, 256, 0, stream>>>(Wk, Wkb, w8);
    cvt_f32_to_bf16<<<w8 / 256, 256, 0, stream>>>(Wv, Wvb, w8);
    cvt_f32_to_bf16<<<w8 / 256, 256, 0, stream>>>(Wo, Wob, w8);
    const int nwords = (int)((size_t)4 * S_LEN * SW);       // 524,288
    pack_mask_bits<<<nwords / 256, 256, 0, stream>>>(mask, Mp, nwords);

    // 2) projections; Q pre-scaled by 1/sqrt(Dk)*log2(e); V stored transposed
    mha_proj<false><<<1024, 256, 0, stream>>>(Xqb, Wqb, bq, Qb, 0.125f * LOG2E);
    mha_proj<false><<<1024, 256, 0, stream>>>(Xkb, Wkb, bk, Kb, 1.0f);
    mha_proj<true ><<<1024, 256, 0, stream>>>(Xvb, Wvb, bv, Vt, 1.0f);

    // 3) flash attention: 4096 waves (32 queries each) / 4 per block
    mha_flash_attn<<<1024, 128, 0, stream>>>(Qb, Kb, Vt, Mp, Hb);

    // 4) output projection -> f32
    mha_out_proj<<<1024, 256, 0, stream>>>(Hb, Wob, bo, (float*)d_out);
}
